// AE_control_54546084659681
// MI455X (gfx1250) — matrix operand order repaired and checked by reading, not compiled, not run on hardware
//
#include <hip/hip_runtime.h>

typedef _Float16 h16;
typedef __attribute__((ext_vector_type(16))) _Float16 v16h;
typedef __attribute__((ext_vector_type(8)))  float    v8f;

#define TLEN 512

// ---------------------------------------------------------------------------
// Weight prep: (Co,Ci,9) f32 -> f16 [co][ci*10+k], zero tap at k==9.
// ---------------------------------------------------------------------------
__global__ void prep_conv_kernel(const float* __restrict__ w, h16* __restrict__ o,
                                 int Co, int Ci)
{
    int i = blockIdx.x * 256 + threadIdx.x;
    int tot = Co * Ci * 10;
    if (i >= tot) return;
    int k = i % 10, rest = i / 10, ci = rest % Ci, co = rest / Ci;
    o[i] = (h16)((k < 9) ? w[((size_t)co * Ci + ci) * 9 + k] : 0.f);
}

__global__ void prep_f16_kernel(const float* __restrict__ s, h16* __restrict__ d, int n)
{
    int i = blockIdx.x * 256 + threadIdx.x;
    if (i < n) d[i] = (h16)s[i];
}

// ---------------------------------------------------------------------------
// enc0 front: 1->64 conv (K=9) + bias + relu  -> h1 (f16)
//             1x1 shortcut conv + bias        -> sc (f16)
// ---------------------------------------------------------------------------
__global__ __launch_bounds__(256)
void enc0_front_kernel(const float* __restrict__ x,  const float* __restrict__ w1,
                       const float* __restrict__ b1, const float* __restrict__ ws,
                       const float* __restrict__ bs,
                       h16* __restrict__ h1, h16* __restrict__ sc)
{
    __shared__ float xl[520];     // t in [-4, 515]
    __shared__ float wl[64 * 9];
    const int b = blockIdx.x, tid = threadIdx.x;
    for (int i = tid; i < 520; i += 256) {
        int t = i - 4;
        xl[i] = (t >= 0 && t < TLEN) ? x[(size_t)b * TLEN + t] : 0.f;
    }
    for (int i = tid; i < 576; i += 256) wl[i] = w1[i];
    __syncthreads();
    for (int i = tid; i < 64 * TLEN; i += 256) {
        int co = i >> 9, t = i & 511;
        float a = b1[co];
#pragma unroll
        for (int k = 0; k < 9; ++k) a += wl[co * 9 + k] * xl[t + k];
        size_t o = ((size_t)b * 64 + co) * TLEN + t;
        h1[o] = (h16)fmaxf(a, 0.f);
        sc[o] = (h16)(ws[co] * xl[t + 4] + bs[co]);
    }
}

// ---------------------------------------------------------------------------
// Implicit-GEMM conv1d (SAME, K=9) via v_wmma_f32_16x16x32_f16.
//   in   : f16 [B][C][T]
//   Wp   : f16 [C][C*10]   (kk = ci*10+k, zero weight at k==9)
//   out  = relu(conv + bias (+ resid))  as f16 [B][C][T]
// Block: one (b, 256-wide t chunk). LDS x tile with 4-left/12-right halo.
// Waves: C*8/32; each wave: co_tile = w % (C/16), four t-tiles of 16.
// ---------------------------------------------------------------------------
template<int C>
__global__ __launch_bounds__(C * 8)
void conv_wmma(const h16* __restrict__ in, const h16* __restrict__ Wp,
               const float* __restrict__ bias, const h16* __restrict__ resid,
               h16* __restrict__ out)
{
    constexpr int KK = C * 10;
    constexpr int XW = 272;                    // 256 + 16 halo
    __shared__ h16 xs[C][XW];
    const int b  = blockIdx.x >> 1;
    const int tc = (blockIdx.x & 1) * 256;
    const h16* xin = in + (size_t)b * C * TLEN;
    const int tid = threadIdx.x;
    constexpr int NTHR = C * 8;

    for (int i = tid; i < C * XW; i += NTHR) {  // exact trip count for all lanes
        int ci = i / XW, u = i - ci * XW;
        int t = tc + u - 4;
        xs[ci][u] = (t >= 0 && t < TLEN) ? xin[ci * TLEN + t] : (h16)0.0f;
    }
    __syncthreads();

    const int wave = tid >> 5, lane = tid & 31;
    const int half = lane >> 4, lm = lane & 15;
    constexpr int COT = C / 16;
    const int cot = wave % COT;
    const int t0  = (wave / COT) * 64;         // 4 t-tiles per wave
    const int co_base = cot * 16;

    v8f acc[4] = {};
    const unsigned* w32base = (const unsigned*)(Wp + (size_t)(co_base + lm) * KK);

    for (int kb = 0; kb < KK; kb += 32) {
        // ---- A fragment (weights, global; ISA 16-bit A 16x32 layout) ----
        v16h af;
#pragma unroll
        for (int j = 0; j < 8; ++j) {
            int koff = (j < 4) ? ((half ? 8 : 0) + 2 * j)
                               : ((half ? 24 : 16) + 2 * (j - 4));
            union { unsigned u; h16 h[2]; } cv;
            cv.u = w32base[(kb + koff) >> 1];
            af[2 * j] = cv.h[0]; af[2 * j + 1] = cv.h[1];
        }
        // ---- decode (ci,k) for this lane's 8 K-pairs ----
        int cis[8], ks[8];
#pragma unroll
        for (int j = 0; j < 8; ++j) {
            int kk = kb + ((j < 4) ? ((half ? 8 : 0) + 2 * j) : ((half ? 24 : 16) + 2 * (j - 4)));
            cis[j] = kk / 10;
            ks[j]  = kk - 10 * cis[j];         // even, pair (k,k+1) adjacent in t
        }
#pragma unroll
        for (int tt = 0; tt < 4; ++tt) {
            int ucol = t0 + tt * 16 + lm;      // column n = lm
            v16h bf;
#pragma unroll
            for (int j = 0; j < 8; ++j) {
                bf[2 * j]     = xs[cis[j]][ucol + ks[j]];
                bf[2 * j + 1] = xs[cis[j]][ucol + ks[j] + 1];
            }
            acc[tt] = __builtin_amdgcn_wmma_f32_16x16x32_f16(
                false, af, false, bf, (short)0, acc[tt], false, false);
        }
    }

    // ---- epilogue: bias (+resid) + relu, store f16 ----
#pragma unroll
    for (int tt = 0; tt < 4; ++tt) {
        int t = tc + t0 + tt * 16 + lm;
#pragma unroll
        for (int r = 0; r < 8; ++r) {
            int co = co_base + r + 8 * half;   // D: M = r + 8*half, N = lm
            float v = acc[tt][r] + bias[co];
            if (resid) v += (float)resid[((size_t)b * C + co) * TLEN + t];
            out[((size_t)b * C + co) * TLEN + t] = (h16)fmaxf(v, 0.f);
        }
    }
}

// ---------------------------------------------------------------------------
// code_assign: softmax-VQ. One thread per (b,t). q -> f16, argmin -> float.
// h is the 64-channel encoder output; ch_off selects code_s / code_n half.
// ---------------------------------------------------------------------------
__global__ __launch_bounds__(128)
void code_assign_kernel(const h16* __restrict__ h, int ch_off,
                        const float* __restrict__ means, int M,
                        h16* __restrict__ q, float* __restrict__ idx_out)
{
    __shared__ float mS[32 * 64];
    __shared__ float mnorm[64];
    const int tid = threadIdx.x;
    for (int i = tid; i < 32 * M; i += 128) mS[i] = means[i];
    __syncthreads();
    if (tid < M) {
        float s = 0.f;
        for (int c = 0; c < 32; ++c) { float v = mS[c * M + tid]; s += v * v; }
        mnorm[tid] = s;
    }
    __syncthreads();

    const int g = blockIdx.x * 128 + tid;
    const int b = g >> 9, t = g & 511;

    float xv[32]; float xn = 0.f;
#pragma unroll
    for (int c = 0; c < 32; ++c) {
        float v = (float)h[((size_t)b * 64 + ch_off + c) * TLEN + t];
        xv[c] = v; xn += v * v;
    }
    float d2[64]; float dmin = 3.4e38f; int am = 0;
    for (int m = 0; m < M; ++m) {
        float dot = 0.f;
#pragma unroll
        for (int c = 0; c < 32; ++c) dot += xv[c] * mS[c * M + m];
        float d = xn - 2.f * dot + mnorm[m];
        d2[m] = d;
        if (d < dmin) { dmin = d; am = m; }
    }
    float ps = 0.f;
    for (int m = 0; m < M; ++m) { float p = __expf(dmin - d2[m]); d2[m] = p; ps += p; }
    float inv = 1.f / ps;
#pragma unroll
    for (int c = 0; c < 32; ++c) {
        float a = 0.f;
        for (int m = 0; m < M; ++m) a += d2[m] * mS[c * M + m];
        q[((size_t)b * 32 + c) * TLEN + t] = (h16)(a * inv);
    }
    idx_out[(size_t)b * 512 + t] = (float)am;
}

// ---------------------------------------------------------------------------
// FC 16384 -> 512 with tanh, WMMA GEMM. M = batch rows, N = outputs, K = 16384.
// grid = (B/16, 512/128); 8 waves, each one 16x16 o-tile.
// ---------------------------------------------------------------------------
__global__ __launch_bounds__(256)
void fc_tanh_kernel(const h16* __restrict__ z,   // [B][16384]
                    const h16* __restrict__ Wf,  // [512][16384]
                    const float* __restrict__ fb,
                    float* __restrict__ out)     // [B][512]
{
    const int tid = threadIdx.x;
    const int wave = tid >> 5, lane = tid & 31;
    const int half = lane >> 4, lm = lane & 15;
    const int bt = blockIdx.x;
    const int o0 = (blockIdx.y * 8 + wave) * 16;

    const unsigned* arow = (const unsigned*)(z  + (size_t)(bt * 16 + lm) * 16384);
    const unsigned* wrow = (const unsigned*)(Wf + (size_t)(o0 + lm) * 16384);

    v8f acc = {};
    for (int kb = 0; kb < 16384; kb += 32) {
        v16h af, bf;
#pragma unroll
        for (int j = 0; j < 8; ++j) {
            int ka = (j < 4) ? ((half ? 8 : 0) + 2 * j)
                             : ((half ? 24 : 16) + 2 * (j - 4));
            union { unsigned u; h16 h[2]; } c1; c1.u = arow[(kb + ka) >> 1];
            af[2 * j] = c1.h[0]; af[2 * j + 1] = c1.h[1];
            int ko = ka;
            union { unsigned u; h16 h[2]; } c2; c2.u = wrow[(kb + ko) >> 1];
            bf[2 * j] = c2.h[0]; bf[2 * j + 1] = c2.h[1];
        }
        acc = __builtin_amdgcn_wmma_f32_16x16x32_f16(
            false, af, false, bf, (short)0, acc, false, false);
    }
#pragma unroll
    for (int r = 0; r < 8; ++r) {
        int bo = bt * 16 + r + 8 * half;
        int o  = o0 + lm;
        out[(size_t)bo * 512 + o] = tanhf(acc[r] + fb[o]);
    }
}

// ---------------------------------------------------------------------------
extern "C" void kernel_launch(void* const* d_in, const int* in_sizes, int n_in,
                              void* d_out, int out_size, void* d_ws, size_t ws_size,
                              hipStream_t stream)
{
    const float* x       = (const float*)d_in[0];
    const float* e0w1    = (const float*)d_in[1];
    const float* e0b1    = (const float*)d_in[2];
    const float* e0w2    = (const float*)d_in[3];
    const float* e0b2    = (const float*)d_in[4];
    const float* e0ws    = (const float*)d_in[5];
    const float* e0bs    = (const float*)d_in[6];
    const float* encw    = (const float*)d_in[7];
    const float* encb    = (const float*)d_in[8];
    const float* decw    = (const float*)d_in[9];
    const float* decb    = (const float*)d_in[10];
    const float* fcw     = (const float*)d_in[11];
    const float* fcb     = (const float*)d_in[12];
    const float* means_s = (const float*)d_in[13];
    const float* means_n = (const float*)d_in[14];
    float* out = (float*)d_out;

    char* ws = (char*)d_ws;
    size_t off = 0;
    auto carve = [&](size_t bytes) {
        char* p = ws + off;
        off = (off + bytes + 255) & ~(size_t)255;
        return p;
    };
    h16* Wp_e0 = (h16*)carve((size_t)64 * 640 * sizeof(h16));
    h16* Wp_enc[6]; for (int i = 0; i < 6; ++i) Wp_enc[i] = (h16*)carve((size_t)64 * 640 * sizeof(h16));
    h16* Wp_dec[6]; for (int i = 0; i < 6; ++i) Wp_dec[i] = (h16*)carve((size_t)32 * 320 * sizeof(h16));
    h16* Wfc = (h16*)carve((size_t)512 * 16384 * sizeof(h16));
    const size_t actB = (size_t)1024 * 64 * 512 * sizeof(h16);   // 64 MB
    h16* R0 = (h16*)carve(actB);
    h16* R1 = (h16*)carve(actB);
    h16* R2 = (h16*)carve(actB);
    h16* R3 = (h16*)carve(actB / 2);                              // 32-channel buffer

    // ---- weight prep (f32 -> padded f16) ----
    prep_conv_kernel<<<160, 256, 0, stream>>>(e0w2, Wp_e0, 64, 64);
    for (int i = 0; i < 6; ++i)
        prep_conv_kernel<<<160, 256, 0, stream>>>(encw + (size_t)i * 64 * 64 * 9, Wp_enc[i], 64, 64);
    for (int i = 0; i < 6; ++i)
        prep_conv_kernel<<<40, 256, 0, stream>>>(decw + (size_t)i * 32 * 32 * 9, Wp_dec[i], 32, 32);
    prep_f16_kernel<<<(512 * 16384 + 255) / 256, 256, 0, stream>>>(fcw, Wfc, 512 * 16384);

    // ---- encoder ----
    enc0_front_kernel<<<1024, 256, 0, stream>>>(x, e0w1, e0b1, e0ws, e0bs, R0, R1);
    conv_wmma<64><<<2048, 512, 0, stream>>>(R0, Wp_e0, e0b2, R1, R2);
    for (int i = 0; i < 3; ++i) {
        conv_wmma<64><<<2048, 512, 0, stream>>>(R2, Wp_enc[2 * i],     encb + (2 * i) * 64,     nullptr, R0);
        conv_wmma<64><<<2048, 512, 0, stream>>>(R0, Wp_enc[2 * i + 1], encb + (2 * i + 1) * 64, R2,      R2);
    }

    // ---- VQ code assignment ----
    code_assign_kernel<<<4096, 128, 0, stream>>>(R2, 0,  means_s, 64, R1, out + 2 * 524288);
    code_assign_kernel<<<4096, 128, 0, stream>>>(R2, 32, means_n, 32, R3, out + 3 * 524288);

    // ---- decoder (s) ----
    for (int i = 0; i < 3; ++i) {
        conv_wmma<32><<<2048, 256, 0, stream>>>(R1, Wp_dec[2 * i],     decb + (2 * i) * 32,     nullptr, R0);
        conv_wmma<32><<<2048, 256, 0, stream>>>(R0, Wp_dec[2 * i + 1], decb + (2 * i + 1) * 32, R1,      R1);
    }
    fc_tanh_kernel<<<dim3(64, 4), 256, 0, stream>>>(R1, Wfc, fcb, out);

    // ---- decoder (n) ----
    for (int i = 0; i < 3; ++i) {
        conv_wmma<32><<<2048, 256, 0, stream>>>(R3, Wp_dec[2 * i],     decb + (2 * i) * 32,     nullptr, R0);
        conv_wmma<32><<<2048, 256, 0, stream>>>(R0, Wp_dec[2 * i + 1], decb + (2 * i + 1) * 32, R3,      R3);
    }
    fc_tanh_kernel<<<dim3(64, 4), 256, 0, stream>>>(R3, Wfc, fcb, out + 524288);

    (void)in_sizes; (void)n_in; (void)out_size; (void)ws_size;
}
